// ReduceLayer_33887291965655
// MI455X (gfx1250) — matrix operand order repaired and checked by reading, not compiled, not run on hardware
//
#include <hip/hip_runtime.h>
#include <hip/hip_bf16.h>
#include <stdint.h>

// ---------------- problem constants ----------------
#define S_TOK    2048
#define DIN      4096
#define DOUT     11008
#define REMAINED 3852          // int(11008 * 0.35)

// ---------------- GEMM tiling ----------------
#define BM 128
#define BN 128
#define BK 32
#define LPITCH 40              // LDS row pitch in halves (32 + 8 pad)

typedef __attribute__((ext_vector_type(16))) _Float16 v16h;
typedef __attribute__((ext_vector_type(8)))  float    v8f;

// =====================================================================
// Kernel 0: zero workspace ints (pos, limit, colCounts, idx)
// =====================================================================
__global__ void zero_ws_kernel(int* __restrict__ p, int n) {
    int i = blockIdx.x * 256 + threadIdx.x;
    if (i < n) p[i] = 0;
}

// =====================================================================
// Kernel 1: tv = x @ W^T + bias   (f16 inputs, f32 out) via WMMA,
//           plus global count of tv > 0 elements.
// Block: 256 threads = 8 waves (2 x 4). Wave tile: 64(M) x 32(N).
// Double-buffered LDS fed by GLOBAL_LOAD_ASYNC_TO_LDS_B128 (no VGPR
// staging, tracked with ASYNCcnt); one barrier per K-step.
// =====================================================================
__global__ __launch_bounds__(256) void gemm_tv_kernel(
    const _Float16* __restrict__ x,      // [S_TOK, DIN]
    const _Float16* __restrict__ w,      // [DOUT, DIN]
    const _Float16* __restrict__ bias,   // [DOUT]
    float*          __restrict__ tv,     // [S_TOK, DOUT]
    unsigned int*   __restrict__ pos)    // scalar counter
{
    __shared__ _Float16 la[2][BM][LPITCH];
    __shared__ _Float16 lb[2][BN][LPITCH];

    const int tid   = threadIdx.x;
    const int lane  = tid & 31;
    const int wave  = tid >> 5;
    const int wm    = wave >> 2;     // 0..1  -> 64-row slab
    const int wn    = wave & 3;      // 0..3  -> 32-col slab
    const int lhalf = lane & 15;     // lane within half-wave
    const int khi   = lane >> 4;     // 0 or 1 (K-half select per ISA layout)
    const int bm0   = blockIdx.y * BM;
    const int bn0   = blockIdx.x * BN;

    // staging coordinates for the 4 x 16B chunks this thread DMAs per tile
    int srow[2], skc[2];
#pragma unroll
    for (int i = 0; i < 2; ++i) {
        const int c = tid + i * 256;     // 0..511
        srow[i] = c >> 2;                // 0..127
        skc[i]  = (c & 3) * 8;           // 0,8,16,24 halves
    }

    // LDS byte offsets of this thread's destination chunks (both buffers).
    // Aperture rule: LDS offset = flat_addr[31:0].
    unsigned lofa[2][2], lofb[2][2];
#pragma unroll
    for (int b = 0; b < 2; ++b)
#pragma unroll
        for (int i = 0; i < 2; ++i) {
            lofa[b][i] = (unsigned)(uintptr_t)&la[b][srow[i]][skc[i]];
            lofb[b][i] = (unsigned)(uintptr_t)&lb[b][srow[i]][skc[i]];
        }

    // async DMA of one 128x32 A tile + 128x32 B tile into buffer b at col k
    auto async_stage = [&](int b, int k) {
#pragma unroll
        for (int i = 0; i < 2; ++i) {
            const unsigned long long ga = (unsigned long long)(uintptr_t)
                (x + (size_t)(bm0 + srow[i]) * DIN + k + skc[i]);
            const unsigned long long gb = (unsigned long long)(uintptr_t)
                (w + (size_t)(bn0 + srow[i]) * DIN + k + skc[i]);
            asm volatile("global_load_async_to_lds_b128 %0, %1, off"
                         :: "v"(lofa[b][i]), "v"(ga) : "memory");
            asm volatile("global_load_async_to_lds_b128 %0, %1, off"
                         :: "v"(lofb[b][i]), "v"(gb) : "memory");
        }
    };

    const v8f zacc = {0.f, 0.f, 0.f, 0.f, 0.f, 0.f, 0.f, 0.f};
    v8f acc[4][2];
#pragma unroll
    for (int i = 0; i < 4; ++i)
#pragma unroll
        for (int j = 0; j < 2; ++j) acc[i][j] = zacc;

    // ---- prologue: DMA tile k=0 into buffer 0 ----
    async_stage(0, 0);
    asm volatile("s_wait_asynccnt 0" ::: "memory");
    __syncthreads();

    int buf = 0;
    for (int k0 = 0; k0 < DIN; k0 += BK, buf ^= 1) {
        const bool has_next = (k0 + BK) < DIN;

        // ---- kick off next tile's DMA first: overlaps with WMMA chain ----
        if (has_next) {
            async_stage(buf ^ 1, k0 + BK);
            if (k0 + 2 * BK < DIN) {   // keep L2 streaming one tile further ahead
                __builtin_prefetch(x + (size_t)(bm0 + (tid >> 1)) * DIN + k0 + 2 * BK, 0, 1);
                __builtin_prefetch(w + (size_t)(bn0 + (tid >> 1)) * DIN + k0 + 2 * BK, 0, 1);
            }
        }

        // ---- load fragments from current buffer ----
        // B fragments (32x16 per tile), ISA wave32 layout:
        // lanes 0-15: column N=lhalf, K in {0..7,16..23} ; lanes 16-31: K in {8..15,24..31}
        v16h bfrag[2];
#pragma unroll
        for (int bt = 0; bt < 2; ++bt) {
            const int n = wn * 32 + bt * 16 + lhalf;
            const _Float16* p = &lb[buf][n][khi * 8];
            uint4* d = (uint4*)&bfrag[bt];
            d[0] = *(const uint4*)p;
            d[1] = *(const uint4*)(p + 16);
        }
        // A fragments (16x32 per tile), ISA wave32 layout:
        // lanes 0-15: K in {0..7,16..23}; lanes 16-31: K in {8..15,24..31}
        v16h afrag[4];
#pragma unroll
        for (int mt = 0; mt < 4; ++mt) {
            const int r = wm * 64 + mt * 16 + lhalf;
            const _Float16* p = &la[buf][r][khi * 8];
            uint4* d = (uint4*)&afrag[mt];
            d[0] = *(const uint4*)p;         // K = khi*8 .. khi*8+7
            d[1] = *(const uint4*)(p + 16);  // K = khi*8+16 .. khi*8+23
        }

        // ---- WMMA chain ----
#pragma unroll
        for (int mt = 0; mt < 4; ++mt)
#pragma unroll
            for (int bt = 0; bt < 2; ++bt)
                acc[mt][bt] = __builtin_amdgcn_wmma_f32_16x16x32_f16(
                    /*neg_a=*/false, afrag[mt], /*neg_b=*/false, bfrag[bt],
                    /*c_mod=*/(short)0, acc[mt][bt],
                    /*reuse_a=*/false, /*reuse_b=*/false);

        // ---- wait for next tile's DMA, then one barrier per K-step ----
        if (has_next)
            asm volatile("s_wait_asynccnt 0" ::: "memory");
        __syncthreads();
    }

    // ---- epilogue: add bias, store f32 tv, count positives ----
    // C/D layout: lane<16 -> N=lane, M=r ; lane>=16 -> N=lane-16, M=r+8
    unsigned int cnt = 0;
#pragma unroll
    for (int mt = 0; mt < 4; ++mt) {
#pragma unroll
        for (int bt = 0; bt < 2; ++bt) {
            const int n = bn0 + wn * 32 + bt * 16 + lhalf;
            const float bv = (float)bias[n];
#pragma unroll
            for (int r = 0; r < 8; ++r) {
                const int m = bm0 + wm * 64 + mt * 16 + r + khi * 8;
                const float v = acc[mt][bt][r] + bv;
                tv[(size_t)m * DOUT + n] = v;
                cnt += (v > 0.0f) ? 1u : 0u;
            }
        }
    }
    atomicAdd(pos, cnt);
}

// =====================================================================
// Kernel 2: limit = floor(TOKEN_SPARSITY * pos / S)
// =====================================================================
__global__ void limit_kernel(const unsigned int* __restrict__ pos,
                             int* __restrict__ limit) {
    if (threadIdx.x == 0 && blockIdx.x == 0) {
        const double p = (double)(*pos);
        *limit = (int)(0.2 * p / (double)S_TOK);
    }
}

// =====================================================================
// Kernel 3: per-row top-`limit` selection via 2048-bucket key histogram;
// selected columns bump colCounts. One block per row.
// =====================================================================
__global__ __launch_bounds__(256) void row_topk_counts_kernel(
    const float* __restrict__ tv,
    const int*   __restrict__ limit_p,
    int*         __restrict__ colCounts)
{
    __shared__ int hist[2048];
    __shared__ unsigned int sBucket;
    const int tid = threadIdx.x;
    const float* rowp = tv + (size_t)blockIdx.x * DOUT;

    for (int i = tid; i < 2048; i += 256) hist[i] = 0;
    __syncthreads();

    for (int i = tid; i < DOUT; i += 256) {
        const unsigned int u = __float_as_uint(rowp[i]);
        const unsigned int key = (u & 0x80000000u) ? ~u : (u | 0x80000000u);
        atomicAdd(&hist[key >> 21], 1);
    }
    __syncthreads();

    if (tid == 0) {
        const int limit = *limit_p;
        unsigned int b = 2048u;   // 2048 => select nothing
        if (limit > 0) {
            int acc = 0;
            for (int i = 2047; i >= 0; --i) {
                acc += hist[i];
                if (acc >= limit) { b = (unsigned int)i; break; }
            }
        }
        sBucket = b;
    }
    __syncthreads();

    const unsigned int bthr = sBucket;
    for (int i = tid; i < DOUT; i += 256) {
        const unsigned int u = __float_as_uint(rowp[i]);
        const unsigned int key = (u & 0x80000000u) ? ~u : (u | 0x80000000u);
        if ((key >> 21) >= bthr) atomicAdd(&colCounts[i], 1);
    }
}

// =====================================================================
// Kernel 4: top-REMAINED columns by count (single block, deterministic
// ordered selection), writes int indices to ws and float idx/bias to out.
// =====================================================================
__global__ __launch_bounds__(256) void select_topk_kernel(
    const int*      __restrict__ colCounts,
    const _Float16* __restrict__ bias,
    int*            __restrict__ idx_ws,
    float*          __restrict__ out_idx,
    float*          __restrict__ out_bias)
{
    __shared__ int hist[2049];
    const int tid = threadIdx.x;
    for (int i = tid; i < 2049; i += 256) hist[i] = 0;
    __syncthreads();

    for (int i = tid; i < DOUT; i += 256) {
        int c = colCounts[i];
        if (c < 0) c = 0;
        if (c > 2048) c = 2048;
        atomicAdd(&hist[c], 1);
    }
    __syncthreads();

    if (tid == 0) {
        int acc = 0, t = 0;
        for (int v = 2048; v >= 0; --v) {
            acc += hist[v];
            if (acc >= REMAINED) { t = v; break; }
        }
        int wpos = 0;
        for (int i = 0; i < DOUT && wpos < REMAINED; ++i)
            if (colCounts[i] > t) idx_ws[wpos++] = i;
        for (int i = 0; i < DOUT && wpos < REMAINED; ++i)
            if (colCounts[i] == t) idx_ws[wpos++] = i;
    }
    __syncthreads();

    for (int i = tid; i < REMAINED; i += 256) {
        const int c = idx_ws[i];
        out_idx[i]  = (float)c;
        out_bias[i] = (float)bias[c];
    }
}

// =====================================================================
// Kernel 5: gather filtered_W rows (f16 -> f32). One block per row.
// =====================================================================
__global__ __launch_bounds__(256) void gather_w_kernel(
    const _Float16* __restrict__ w,
    const int*      __restrict__ idx_ws,
    float*          __restrict__ outW)
{
    const int r = blockIdx.x;
    const int src = idx_ws[r];
    const _Float16* wp = w + (size_t)src * DIN;
    float* op = outW + (size_t)r * DIN;
    for (int j = threadIdx.x; j < DIN; j += 256) op[j] = (float)wp[j];
}

// =====================================================================
extern "C" void kernel_launch(void* const* d_in, const int* in_sizes, int n_in,
                              void* d_out, int out_size, void* d_ws, size_t ws_size,
                              hipStream_t stream) {
    (void)in_sizes; (void)n_in; (void)out_size; (void)ws_size;

    const _Float16* x    = (const _Float16*)d_in[0];   // [2048, 4096] f16
    const _Float16* w    = (const _Float16*)d_in[1];   // [11008, 4096] f16
    const _Float16* bias = (const _Float16*)d_in[2];   // [11008] f16

    float* out   = (float*)d_out;
    float* tv    = out;                                   // S*DOUT
    float* outW  = tv   + (size_t)S_TOK * DOUT;           // REMAINED*DIN
    float* outB  = outW + (size_t)REMAINED * DIN;         // REMAINED
    float* outI  = outB + REMAINED;                       // REMAINED

    int* wsi = (int*)d_ws;
    unsigned int* pos = (unsigned int*)wsi;               // [0]
    int* limit        = wsi + 1;                          // [1]
    int* colCounts    = wsi + 2;                          // [2 .. 2+DOUT)
    int* idx_ws       = wsi + 2 + DOUT;                   // [.. +REMAINED)

    const int nzero = 2 + DOUT + REMAINED;
    zero_ws_kernel<<<(nzero + 255) / 256, 256, 0, stream>>>(wsi, nzero);

    dim3 grid(DOUT / BN, S_TOK / BM);                     // 86 x 16
    gemm_tv_kernel<<<grid, 256, 0, stream>>>(x, w, bias, tv, pos);

    limit_kernel<<<1, 1, 0, stream>>>(pos, limit);
    row_topk_counts_kernel<<<S_TOK, 256, 0, stream>>>(tv, limit, colCounts);
    select_topk_kernel<<<1, 256, 0, stream>>>(colCounts, bias, idx_ws, outI, outB);
    gather_w_kernel<<<REMAINED, 256, 0, stream>>>(w, idx_ws, outW);
}
